// AttnDecoder_61125974556734
// MI455X (gfx1250) — matrix operand order repaired and checked by reading, not compiled, not run on hardware
//
#include <hip/hip_runtime.h>
#include <hip/hip_bf16.h>

// Problem constants (from reference)
#define DIMD 512
#define DICTN 32000
#define BB   16
#define TT   64
#define SS   64
#define G4   2048   // 4*DIM

typedef __bf16 bf16_t;
typedef __attribute__((ext_vector_type(16))) __bf16 v16bf;
typedef __attribute__((ext_vector_type(8)))  float  v8f;

#define WMMA_BF16(a, b, c) \
  __builtin_amdgcn_wmma_f32_16x16x32_bf16(false, (a), false, (b), (short)0, (c), false, false)

// ---------------------------------------------------------------------------
// WMMA fragment helpers (ISA 7.12.2 layouts, wave32)
// A: 16x32 bf16. lane&15 = M row; lanes 0-15 hold K {k0+0..7, k0+16..23},
//    lanes 16-31 hold K {k0+8..15, k0+24..31}.
// ---------------------------------------------------------------------------
__device__ __forceinline__ v16bf load_a_bf16(const bf16_t* src, int lda, int k0) {
  const int lane = threadIdx.x & 31;
  const int m    = lane & 15;
  const int kb   = k0 + ((lane & 16) ? 8 : 0);
  const bf16_t* p = src + (size_t)m * lda + kb;
  v16bf a;
#pragma unroll
  for (int i = 0; i < 8; ++i) { a[i] = p[i]; a[8 + i] = p[16 + i]; }
  return a;
}

__device__ __forceinline__ v16bf load_a_f32cvt(const float* src, int lda, int k0) {
  const int lane = threadIdx.x & 31;
  const int m    = lane & 15;
  const int kb   = k0 + ((lane & 16) ? 8 : 0);
  const float* p = src + (size_t)m * lda + kb;
  v16bf a;
#pragma unroll
  for (int i = 0; i < 8; ++i) { a[i] = (bf16_t)p[i]; a[8 + i] = (bf16_t)p[16 + i]; }
  return a;
}

// B fragment from weights PACKED in fragment order: packed[(frag*32+lane)*16 + i].
// One wave reads one contiguous 1KB block -> perfectly coalesced.
__device__ __forceinline__ v16bf load_b_pk(const bf16_t* packed, int frag) {
  const int lane = threadIdx.x & 31;
  const bf16_t* p = packed + ((size_t)frag * 32 + lane) * 16;
  v16bf b;
#pragma unroll
  for (int i = 0; i < 16; ++i) b[i] = p[i];
  return b;
}

// C/D 16x16 f32: lane&15 = N; vgpr r holds M = r + (lane>=16 ? 8 : 0)
__device__ __forceinline__ void store_d_f32(float* dst, int ldd, v8f c) {
  const int lane = threadIdx.x & 31;
  const int n    = lane & 15;
  const int mb   = (lane & 16) ? 8 : 0;
#pragma unroll
  for (int r = 0; r < 8; ++r) dst[(size_t)(mb + r) * ldd + n] = c[r];
}

__device__ __forceinline__ float sigm(float x) { return 1.0f / (1.0f + __expf(-x)); }

// ---------------------------------------------------------------------------
// One-time kernels
// ---------------------------------------------------------------------------
// Pack W [N x K] row-major f32 -> WMMA-B fragment-packed bf16:
// packed[((ntile*KC+kc)*32 + lane)*16 + i] =
//   (bf16) W[ntile*16 + (lane&15)][kc*32 + ((lane&16)?8:0) + (i<8 ? i : i+8)]
__global__ void ad_pack_b(const float* __restrict__ W, bf16_t* __restrict__ packed,
                          int total, int KC, int K) {
  int tid = blockIdx.x * blockDim.x + threadIdx.x;    // one thread per (frag,lane)
  if (tid >= total) return;
  int lane = tid & 31;
  int frag = tid >> 5;
  int kc = frag % KC, ntile = frag / KC;
  int n = ntile * 16 + (lane & 15);
  int k = kc * 32 + ((lane & 16) ? 8 : 0);
  const float* src = W + (size_t)n * K + k;
  bf16_t* dst = packed + (size_t)tid * 16;
#pragma unroll
  for (int i = 0; i < 16; ++i) dst[i] = (bf16_t)src[(i < 8) ? i : i + 8];
}

// h0 = pre_h[input_len[b]-1, b, :]; c = cell0
__global__ void ad_init(const float* __restrict__ pre_h, const float* __restrict__ cell0,
                        const int* __restrict__ input_len,
                        float* __restrict__ h, float* __restrict__ c,
                        bf16_t* __restrict__ h_bf) {
  int idx = blockIdx.x * blockDim.x + threadIdx.x;           // 0..8191
  int b = idx >> 9, d = idx & 511;
  float h0 = pre_h[((size_t)(input_len[b] - 1) * BB + b) * DIMD + d];
  h[idx] = h0;
  h_bf[idx] = (bf16_t)h0;
  c[idx] = cell0[idx];
}

// xW[t,b,:] = emb[words[b,t]] @ W_ih^T. 8192 tiles, 2 per wave, grid 512 x 256.
__global__ void ad_xw(const int* __restrict__ words, const float* __restrict__ emb,
                      const bf16_t* __restrict__ Wih_pk, float* __restrict__ xW) {
  int wt = blockIdx.x * 8 + (threadIdx.x >> 5);      // 0..4095
  int t = wt >> 6;
  int t0 = (wt & 63) * 2, t1 = t0 + 1;               // two consecutive N-tiles
  const int lane = threadIdx.x & 31;
  const int m = lane & 15;
  const int kboff = (lane & 16) ? 8 : 0;
  const float* arow = emb + (size_t)words[m * TT + t] * DIMD;
  v8f c0 = {}, c1 = {};
#pragma unroll 4
  for (int kc = 0; kc < 16; ++kc) {
    const float* p = arow + kc * 32 + kboff;
    v16bf a;
#pragma unroll
    for (int i = 0; i < 8; ++i) { a[i] = (bf16_t)p[i]; a[8 + i] = (bf16_t)p[16 + i]; }
    c0 = WMMA_BF16(a, load_b_pk(Wih_pk, t0 * 16 + kc), c0);
    c1 = WMMA_BF16(a, load_b_pk(Wih_pk, t1 * 16 + kc), c1);
  }
  float* base = xW + (size_t)t * BB * G4;
  store_d_f32(base + t0 * 16, G4, c0);
  store_d_f32(base + t1 * 16, G4, c1);
}

// Wh_pre[s,b,:] = pre_h[s,b,:] @ W_h^T (1024 rows). 2048 tiles, 2/wave, grid 128.
__global__ void ad_whpre(const float* __restrict__ pre_h, const bf16_t* __restrict__ Wh_pk,
                         float* __restrict__ Wh_pre) {
  int wt = blockIdx.x * 8 + (threadIdx.x >> 5);      // 0..1023
  int row0 = (wt >> 4) * 16;
  int t0 = (wt & 15) * 2, t1 = t0 + 1;
  v8f c0 = {}, c1 = {};
#pragma unroll 4
  for (int kc = 0; kc < 16; ++kc) {
    v16bf a = load_a_f32cvt(pre_h + (size_t)row0 * DIMD, DIMD, kc * 32);
    c0 = WMMA_BF16(a, load_b_pk(Wh_pk, t0 * 16 + kc), c0);
    c1 = WMMA_BF16(a, load_b_pk(Wh_pk, t1 * 16 + kc), c1);
  }
  float* base = Wh_pre + (size_t)row0 * DIMD;
  store_d_f32(base + t0 * 16, DIMD, c0);
  store_d_f32(base + t1 * 16, DIMD, c1);
}

// ---------------------------------------------------------------------------
// Per-step kernels
// ---------------------------------------------------------------------------
// gates g = xW[t] + h @ W_hh^T + (b_ih+b_hh). 128 tiles, 2/wave, grid 8 x 256.
// Software-pipelined: A/B fragments for kc+1 prefetched during kc's WMMAs.
__global__ void ad_gates(const bf16_t* __restrict__ h_bf, const bf16_t* __restrict__ Whh_pk,
                         const float* __restrict__ xW_t, const float* __restrict__ b_ih,
                         const float* __restrict__ b_hh, float* __restrict__ g) {
  int wt = blockIdx.x * 8 + (threadIdx.x >> 5);      // 0..63
  int t0 = wt * 2, t1 = t0 + 1;
  const int f0 = t0 * 16;                            // fragment base (t1 base = f0+16)
  const int lane = threadIdx.x & 31;
  const int n = lane & 15;
  const int mb = (lane & 16) ? 8 : 0;
  int col0 = t0 * 16 + n, col1 = t1 * 16 + n;
  float bias0 = b_ih[col0] + b_hh[col0];
  float bias1 = b_ih[col1] + b_hh[col1];
  v8f c0, c1;
#pragma unroll
  for (int r = 0; r < 8; ++r) {
    c0[r] = xW_t[(size_t)(mb + r) * G4 + col0] + bias0;
    c1[r] = xW_t[(size_t)(mb + r) * G4 + col1] + bias1;
  }
  v16bf a  = load_a_bf16(h_bf, DIMD, 0);
  v16bf b0 = load_b_pk(Whh_pk, f0);
  v16bf b1 = load_b_pk(Whh_pk, f0 + 16);
#pragma unroll
  for (int kc = 0; kc < 16; ++kc) {
    v16bf an, b0n, b1n;
    if (kc < 15) {
      an  = load_a_bf16(h_bf, DIMD, (kc + 1) * 32);
      b0n = load_b_pk(Whh_pk, f0 + kc + 1);
      b1n = load_b_pk(Whh_pk, f0 + 16 + kc + 1);
    }
    c0 = WMMA_BF16(a, b0, c0);
    c1 = WMMA_BF16(a, b1, c1);
    if (kc < 15) { a = an; b0 = b0n; b1 = b1n; }
  }
  store_d_f32(g + t0 * 16, G4, c0);
  store_d_f32(g + t1 * 16, G4, c1);
}

// LSTM pointwise update (8192 elems, grid 32 x 256)
__global__ void ad_lstm(const float* __restrict__ g, float* __restrict__ h,
                        float* __restrict__ c, bf16_t* __restrict__ h_bf,
                        bf16_t* __restrict__ c_bf, bf16_t* __restrict__ ctx_c_bf) {
  int idx = blockIdx.x * blockDim.x + threadIdx.x;
  int b = idx >> 9, d = idx & 511;
  const float* gr = g + (size_t)b * G4;
  float gi = sigm(gr[d]);
  float gf = sigm(gr[512 + d]);
  float gg = tanhf(gr[1024 + d]);
  float go = sigm(gr[1536 + d]);
  float cn = gf * c[idx] + gi * gg;
  float hn = go * tanhf(cn);
  c[idx] = cn;  h[idx] = hn;
  h_bf[idx] = (bf16_t)hn;
  c_bf[idx] = (bf16_t)cn;
  ctx_c_bf[(size_t)b * (2 * DIMD) + DIMD + d] = (bf16_t)cn;   // second half of [ctx|c2]
}

// cW = c2 @ W_s^T + b_s. 32 tiles, 2/wave, grid 2 x 256.
__global__ void ad_cw(const bf16_t* __restrict__ c_bf, const bf16_t* __restrict__ Ws_pk,
                      const float* __restrict__ b_s, float* __restrict__ cW) {
  int wt = blockIdx.x * 8 + (threadIdx.x >> 5);      // 0..15
  int t0 = wt * 2, t1 = t0 + 1;
  const int lane = threadIdx.x & 31;
  const int n = lane & 15;
  float bv0 = b_s[t0 * 16 + n], bv1 = b_s[t1 * 16 + n];
  v8f c0, c1;
#pragma unroll
  for (int r = 0; r < 8; ++r) { c0[r] = bv0; c1[r] = bv1; }
#pragma unroll 4
  for (int kc = 0; kc < 16; ++kc) {
    v16bf a = load_a_bf16(c_bf, DIMD, kc * 32);
    c0 = WMMA_BF16(a, load_b_pk(Ws_pk, t0 * 16 + kc), c0);
    c1 = WMMA_BF16(a, load_b_pk(Ws_pk, t1 * 16 + kc), c1);
  }
  store_d_f32(cW + t0 * 16, DIMD, c0);
  store_d_f32(cW + t1 * 16, DIMD, c1);
}

// e[s,b] = v . tanh(Wh_pre[s,b,:] + cW[b,:])  (1024 rows; 1 wave/row; grid 128 x 256)
__global__ void ad_attn_e(const float* __restrict__ Wh_pre, const float* __restrict__ cW,
                          const float* __restrict__ v_t, float* __restrict__ e) {
  int row = blockIdx.x * 8 + (threadIdx.x >> 5);   // row = s*16 + b
  const int lane = threadIdx.x & 31;
  const float* wp  = Wh_pre + (size_t)row * DIMD;
  const float* cwp = cW + (size_t)(row & 15) * DIMD;
  float sum = 0.f;
#pragma unroll 4
  for (int d = lane; d < DIMD; d += 32) sum += tanhf(wp[d] + cwp[d]) * v_t[d];
#pragma unroll
  for (int off = 16; off > 0; off >>= 1) sum += __shfl_xor(sum, off, 32);
  if (lane == 0) e[row] = sum;
}

// per-b masked softmax over S, then ctx[b,:] = sum_s a[s]*pre_h[s,b,:]  (grid 16 x 256)
__global__ void ad_soft_ctx(const float* __restrict__ e, const int* __restrict__ input_len,
                            const float* __restrict__ pre_h, bf16_t* __restrict__ ctx_c_bf) {
  __shared__ float sa[SS];
  const int b = blockIdx.x;
  const int tid = threadIdx.x;
  if (tid < SS) sa[tid] = (tid < input_len[b]) ? e[tid * BB + b] : -1e9f;
  __syncthreads();
  if (tid == 0) {
    float mx = -1e30f;
    for (int s = 0; s < SS; ++s) mx = fmaxf(mx, sa[s]);
    float sum = 0.f;
    for (int s = 0; s < SS; ++s) { float w = __expf(sa[s] - mx); sa[s] = w; sum += w; }
    float inv = 1.f / sum;
    for (int s = 0; s < SS; ++s) sa[s] *= inv;
  }
  __syncthreads();
  for (int d = tid; d < DIMD; d += 256) {
    float acc = 0.f;
#pragma unroll 8
    for (int s = 0; s < SS; ++s) acc += sa[s] * pre_h[((size_t)s * BB + b) * DIMD + d];
    ctx_c_bf[(size_t)b * (2 * DIMD) + d] = (bf16_t)acc;      // first half of [ctx|c2]
  }
}

// hid = [ctx|c2] @ V^T + b_V -> bf16. 32 tiles, 2/wave, grid 2 x 256. K=1024.
__global__ void ad_hid(const bf16_t* __restrict__ ctx_c_bf, const bf16_t* __restrict__ V_pk,
                       const float* __restrict__ b_V, bf16_t* __restrict__ hid_bf) {
  int wt = blockIdx.x * 8 + (threadIdx.x >> 5);      // 0..15
  int t0 = wt * 2, t1 = t0 + 1;
  const int lane = threadIdx.x & 31;
  const int n = lane & 15;
  const int mb = (lane & 16) ? 8 : 0;
  float bv0 = b_V[t0 * 16 + n], bv1 = b_V[t1 * 16 + n];
  v8f c0, c1;
#pragma unroll
  for (int r = 0; r < 8; ++r) { c0[r] = bv0; c1[r] = bv1; }
#pragma unroll 4
  for (int kc = 0; kc < 32; ++kc) {
    v16bf a = load_a_bf16(ctx_c_bf, 2 * DIMD, kc * 32);
    c0 = WMMA_BF16(a, load_b_pk(V_pk, t0 * 32 + kc), c0);
    c1 = WMMA_BF16(a, load_b_pk(V_pk, t1 * 32 + kc), c1);
  }
#pragma unroll
  for (int r = 0; r < 8; ++r) {
    hid_bf[(size_t)(mb + r) * DIMD + t0 * 16 + n] = (bf16_t)c0[r];
    hid_bf[(size_t)(mb + r) * DIMD + t1 * 16 + n] = (bf16_t)c1[r];
  }
}

// out[t] = hid @ Vp^T + b_Vp, masked by (t < lengths[b]).
// 2000 tiles, 2/wave, grid 125 x 256. A staged in LDS via async copy.
// Inner loop software-pipelined: next K-chunk's A (LDS) and B (global) fragments
// are prefetched while the current pair of WMMAs executes.
#define LDS_LDA 520   // 512 + 8 bf16 pad (1040 B row stride: 16B aligned, bank-spread)
__global__ void ad_proj(const bf16_t* __restrict__ hid_bf, const bf16_t* __restrict__ Vp_pk,
                        const float* __restrict__ b_Vp, const int* __restrict__ lengths,
                        float* __restrict__ out, int t) {
  __shared__ bf16_t sA[16 * LDS_LDA];
  // Stage hid (16x512 bf16 = 16KB) into LDS with async global->LDS b128 copies.
  {
    const int tid = threadIdx.x;
#pragma unroll
    for (int i = 0; i < 4; ++i) {
      int cid = tid + 256 * i;                 // 1024 chunks of 16B
      int row = cid >> 6;
      int cc  = cid & 63;                      // 16B chunk within row
      unsigned laddr = (unsigned)(uintptr_t)sA + (unsigned)(row * (LDS_LDA * 2) + cc * 16);
      unsigned long long gaddr =
          (unsigned long long)(uintptr_t)(hid_bf + (size_t)row * DIMD + cc * 8);
      asm volatile("global_load_async_to_lds_b128 %0, %1, off"
                   :: "v"(laddr), "v"(gaddr) : "memory");
    }
    asm volatile("s_wait_asynccnt 0" ::: "memory");
  }
  __syncthreads();

  int wt = blockIdx.x * 8 + (threadIdx.x >> 5);      // 0..999
  int t0 = wt * 2, t1 = t0 + 1;
  const int f0 = t0 * 16;                            // fragment base (t1 base = f0+16)
  const int lane = threadIdx.x & 31;
  const int n = lane & 15;
  const int mb = (lane & 16) ? 8 : 0;
  const int kboff = (lane & 16) ? 8 : 0;
  float bv0 = b_Vp[t0 * 16 + n], bv1 = b_Vp[t1 * 16 + n];
  v8f c0, c1;
#pragma unroll
  for (int r = 0; r < 8; ++r) { c0[r] = bv0; c1[r] = bv1; }
  const bf16_t* arow = sA + (size_t)(lane & 15) * LDS_LDA + kboff;

  // A fragment from LDS at K-chunk kc
  auto lda_frag = [&](int kc) {
    const bf16_t* p = arow + kc * 32;
    v16bf a;
#pragma unroll
    for (int i = 0; i < 8; ++i) { a[i] = p[i]; a[8 + i] = p[16 + i]; }
    return a;
  };

  v16bf a  = lda_frag(0);
  v16bf b0 = load_b_pk(Vp_pk, f0);
  v16bf b1 = load_b_pk(Vp_pk, f0 + 16);
#pragma unroll
  for (int kc = 0; kc < 16; ++kc) {
    v16bf an, b0n, b1n;
    if (kc < 15) {
      an  = lda_frag(kc + 1);
      b0n = load_b_pk(Vp_pk, f0 + kc + 1);
      b1n = load_b_pk(Vp_pk, f0 + 16 + kc + 1);
    }
    c0 = WMMA_BF16(a, b0, c0);
    c1 = WMMA_BF16(a, b1, c1);
    if (kc < 15) { a = an; b0 = b0n; b1 = b1n; }
  }
  float* ob = out + (size_t)t * BB * DICTN;
#pragma unroll
  for (int r = 0; r < 8; ++r) {
    int m = mb + r;
    bool on = (t < lengths[m]);
    ob[(size_t)m * DICTN + t0 * 16 + n] = on ? c0[r] : 0.f;
    ob[(size_t)m * DICTN + t1 * 16 + n] = on ? c1[r] : 0.f;
  }
}

// ---------------------------------------------------------------------------
// Host launcher
// ---------------------------------------------------------------------------
extern "C" void kernel_launch(void* const* d_in, const int* in_sizes, int n_in,
                              void* d_out, int out_size, void* d_ws, size_t ws_size,
                              hipStream_t stream) {
  const int*   words     = (const int*)  d_in[0];
  const int*   lengths   = (const int*)  d_in[1];
  const int*   input_len = (const int*)  d_in[2];
  const float* pre_h     = (const float*)d_in[3];
  const float* cell0     = (const float*)d_in[4];
  const float* emb       = (const float*)d_in[5];
  const float* W_ih      = (const float*)d_in[6];
  const float* W_hh      = (const float*)d_in[7];
  const float* b_ih      = (const float*)d_in[8];
  const float* b_hh      = (const float*)d_in[9];
  const float* W_h       = (const float*)d_in[10];
  const float* W_s       = (const float*)d_in[11];
  const float* b_s       = (const float*)d_in[12];
  const float* v_t       = (const float*)d_in[13];
  const float* V         = (const float*)d_in[14];
  const float* b_V       = (const float*)d_in[15];
  const float* Vp        = (const float*)d_in[16];
  const float* b_Vp      = (const float*)d_in[17];
  float* out = (float*)d_out;

  // workspace carve-up
  size_t off = 0;
  auto carve = [&](size_t bytes) -> void* {
    void* p = (char*)d_ws + off;
    off = (off + bytes + 255) & ~(size_t)255;
    return p;
  };
  bf16_t* Vp_pk    = (bf16_t*)carve((size_t)DICTN * DIMD * 2);
  bf16_t* Wih_pk   = (bf16_t*)carve((size_t)G4 * DIMD * 2);
  bf16_t* Whh_pk   = (bf16_t*)carve((size_t)G4 * DIMD * 2);
  bf16_t* Wh_pk    = (bf16_t*)carve((size_t)DIMD * DIMD * 2);
  bf16_t* Ws_pk    = (bf16_t*)carve((size_t)DIMD * DIMD * 2);
  bf16_t* V_pk     = (bf16_t*)carve((size_t)DIMD * 2 * DIMD * 2);
  float*  xW       = (float*) carve((size_t)TT * BB * G4 * 4);
  float*  Wh_pre   = (float*) carve((size_t)SS * BB * DIMD * 4);
  float*  g        = (float*) carve((size_t)BB * G4 * 4);
  float*  h        = (float*) carve((size_t)BB * DIMD * 4);
  float*  c        = (float*) carve((size_t)BB * DIMD * 4);
  bf16_t* h_bf     = (bf16_t*)carve((size_t)BB * DIMD * 2);
  bf16_t* c_bf     = (bf16_t*)carve((size_t)BB * DIMD * 2);
  bf16_t* ctx_c_bf = (bf16_t*)carve((size_t)BB * 2 * DIMD * 2);
  float*  cW       = (float*) carve((size_t)BB * DIMD * 4);
  float*  e        = (float*) carve((size_t)SS * BB * 4);
  bf16_t* hid_bf   = (bf16_t*)carve((size_t)BB * DIMD * 2);
  (void)ws_size; (void)n_in; (void)in_sizes; (void)out_size;

  // one-time: pack weights into WMMA fragment order (bf16)
  auto pack = [&](const float* W, bf16_t* P, int Ntiles, int KC, int K) {
    int total = Ntiles * KC * 32;
    ad_pack_b<<<(total + 255) / 256, 256, 0, stream>>>(W, P, total, KC, K);
  };
  pack(Vp,   Vp_pk,  DICTN / 16, 16, DIMD);       // 2000 x 16
  pack(W_ih, Wih_pk, G4 / 16,    16, DIMD);       // 128 x 16
  pack(W_hh, Whh_pk, G4 / 16,    16, DIMD);
  pack(W_h,  Wh_pk,  DIMD / 16,  16, DIMD);       // 32 x 16
  pack(W_s,  Ws_pk,  DIMD / 16,  16, DIMD);
  pack(V,    V_pk,   DIMD / 16,  32, 2 * DIMD);   // 32 x 32

  ad_init<<<32, 256, 0, stream>>>(pre_h, cell0, input_len, h, c, h_bf);
  ad_xw<<<512, 256, 0, stream>>>(words, emb, Wih_pk, xW);
  ad_whpre<<<128, 256, 0, stream>>>(pre_h, Wh_pk, Wh_pre);

  for (int t = 0; t < TT; ++t) {
    ad_gates<<<8, 256, 0, stream>>>(h_bf, Whh_pk, xW + (size_t)t * BB * G4, b_ih, b_hh, g);
    ad_lstm<<<32, 256, 0, stream>>>(g, h, c, h_bf, c_bf, ctx_c_bf);
    ad_cw<<<2, 256, 0, stream>>>(c_bf, Ws_pk, b_s, cW);
    ad_attn_e<<<128, 256, 0, stream>>>(Wh_pre, cW, v_t, e);
    ad_soft_ctx<<<16, 256, 0, stream>>>(e, input_len, pre_h, ctx_c_bf);
    ad_hid<<<2, 256, 0, stream>>>(ctx_c_bf, V_pk, b_V, hid_bf);
    ad_proj<<<125, 256, 0, stream>>>(hid_bf, Vp_pk, b_Vp, lengths, out, t);
  }
}
